// NeuromorphicBrainZone_52415780880478
// MI455X (gfx1250) — compile-verified, matrix operand order checked by reading, not run on hardware
//
#include <hip/hip_runtime.h>
#include <math.h>
#include <stdint.h>

// ---------------------------------------------------------------------------
// NeuromorphicBrainZone for MI455X (gfx1250, wave32)
//
// out[t,j] = b[j] - mean_i |x[t,i] - W[j,i]|
// Exact rewrite:  mean_i |x-w| = (Sx[t] + Sw[j] - 2*Sum_i min(x,w)) / D
//  - min-sum inner loop: v_min_f32 + v_add_f32 (VOP2, dual-issue friendly)
//  - Sx/Sw row sums: V_WMMA_F32_16X16X32_F16 vs. a ones matrix
//  - tile staging: Tensor Data Mover (tensor_load_to_lds), double buffered,
//    tracked with TENSORcnt -> zero staging VALU in the hot loop.
// ---------------------------------------------------------------------------

typedef __attribute__((ext_vector_type(16))) _Float16 v16h;
typedef __attribute__((ext_vector_type(8)))  float    v8f;
typedef __attribute__((ext_vector_type(4)))  unsigned int u32x4;
typedef __attribute__((ext_vector_type(8)))  int      i32x8;
typedef __attribute__((ext_vector_type(4)))  int      i32x4;

#define TT 64          // t-tile rows
#define TJ 64          // j-tile rows
#define KC 32          // K depth staged per TDM transfer
#define LSTR 36        // LDS row stride in floats (32 data + 4 TDM pad)

// ---------------------------------------------------------------------------
// Row-sum of [M x K] f32 via WMMA: out[m] = sum_k in[m,k].  K % 32 == 0.
// One wave per 16 rows; block = 8 waves = 128 rows.
// ---------------------------------------------------------------------------
__global__ __launch_bounds__(256)
void rowsum_wmma_kernel(const float* __restrict__ in, int K,
                        float* __restrict__ out) {
    const int lane = threadIdx.x & 31;
    const int wave = threadIdx.x >> 5;
    const int m    = lane & 15;
    const int h    = lane >> 4;
    const int row0 = blockIdx.x * 128 + wave * 16;
    const float* rp = in + (size_t)(row0 + m) * (size_t)K;

    v16h ones;
#pragma unroll
    for (int i = 0; i < 16; ++i) ones[i] = (_Float16)1.0f;

    v8f c = {};
    for (int k0 = 0; k0 < K; k0 += 32) {
        // A(16x32 f16): lanes 0-15 hold K 0..7,16..23; lanes 16-31 K 8..15,24..31
        const float4 f0 = *(const float4*)(rp + k0 + 8 * h);
        const float4 f1 = *(const float4*)(rp + k0 + 8 * h + 4);
        const float4 f2 = *(const float4*)(rp + k0 + 16 + 8 * h);
        const float4 f3 = *(const float4*)(rp + k0 + 16 + 8 * h + 4);
        v16h a;
        a[0]  = (_Float16)f0.x; a[1]  = (_Float16)f0.y;
        a[2]  = (_Float16)f0.z; a[3]  = (_Float16)f0.w;
        a[4]  = (_Float16)f1.x; a[5]  = (_Float16)f1.y;
        a[6]  = (_Float16)f1.z; a[7]  = (_Float16)f1.w;
        a[8]  = (_Float16)f2.x; a[9]  = (_Float16)f2.y;
        a[10] = (_Float16)f2.z; a[11] = (_Float16)f2.w;
        a[12] = (_Float16)f3.x; a[13] = (_Float16)f3.y;
        a[14] = (_Float16)f3.z; a[15] = (_Float16)f3.w;
        c = __builtin_amdgcn_wmma_f32_16x16x32_f16(
                false, a, false, ones, (short)0, c, false, false);
    }
    if (m == 0) {
#pragma unroll
        for (int r = 0; r < 8; ++r) out[row0 + 8 * h + r] = c[r];
    }
}

// ---------------------------------------------------------------------------
// TDM: DMA one [64 rows x 32 f32] tile (row stride = stride_elems in global)
// into LDS at byte offset lds_byte_off, padding each 32-DWORD row with
// 4 DWORDs -> LDS row stride 36 DWORDs (144 B, 16B aligned, bank-spread).
// 2D descriptor; groups 2/3 zero (tile_dim2 = 0 -> unused).
// ---------------------------------------------------------------------------
__device__ static inline void tdm_load_tile(unsigned lds_byte_off,
                                            const float* gptr,
                                            unsigned stride_elems) {
    u32x4 g0;
    g0[0] = 1u;                                   // count=1, user descriptor
    g0[1] = lds_byte_off;                         // LDS destination (bytes)
    const uint64_t ga = (uint64_t)(uintptr_t)gptr;
    g0[2] = (unsigned)ga;                         // global_addr[31:0]
    g0[3] = (unsigned)((ga >> 32) & 0x01FFFFFFu)  // global_addr[56:32]
          | (2u << 30);                           // type = 2 ("image")

    const unsigned dim0  = stride_elems;          // tensor dim0 (>= tile)
    const unsigned dim1  = TT;                    // 64 rows
    const unsigned tile0 = KC;                    // 32 elements, contiguous
    const unsigned tile1 = TT;                    // 64 rows

    i32x8 g1;
    g1[0] = (int)((2u << 16)        // data_size = 2 -> 4 bytes
                | (1u << 20)        // pad_enable
                | (4u << 22)        // pad_interval: 32 DWORDs
                | (3u << 25));      // pad_amount:   4 DWORDs
    g1[1] = (int)((dim0 & 0xFFFFu) << 16);                 // dim0[15:0]
    g1[2] = (int)((dim0 >> 16) | ((dim1 & 0xFFFFu) << 16));// dim0[31:16],dim1[15:0]
    g1[3] = (int)((dim1 >> 16) | (tile0 << 16));           // dim1[31:16],tile_dim0
    g1[4] = (int)tile1;                                    // tile_dim1 (tile_dim2=0)
    g1[5] = (int)stride_elems;                             // tensor_dim0_stride lo
    g1[6] = 0;                                             // stride hi, dim1_stride lo
    g1[7] = 0;

    i32x4 z4; z4[0] = 0; z4[1] = 0; z4[2] = 0; z4[3] = 0;
#if defined(__clang_major__) && (__clang_major__ >= 23)
    i32x8 z8;
#pragma unroll
    for (int i = 0; i < 8; ++i) z8[i] = 0;
    __builtin_amdgcn_tensor_load_to_lds(g0, g1, z4, z4, z8, 0);
#else
    __builtin_amdgcn_tensor_load_to_lds(g0, g1, z4, z4, 0);
#endif
}

__device__ static inline unsigned lds_off_of(const void* p) {
    // Generic LDS addresses carry the LDS offset in the low 32 bits
    return (unsigned)(uintptr_t)p;
}

// ---------------------------------------------------------------------------
// Fused AdditionLinear (min-sum), TDM double-buffered tiles:
//   acc[t,j] = sum_i min(X[t,i], W[j,i])
//   res      = bias[j] - (Sx[t] + Sw[j] - 2*acc)/D ; SPIKE -> sigmoid(4(res-vth))
// ---------------------------------------------------------------------------
template <bool SPIKE>
__global__ __launch_bounds__(256)
void addmin_kernel(const float* __restrict__ X, const float* __restrict__ W,
                   const float* __restrict__ bias, const float* __restrict__ vth,
                   const float* __restrict__ Sx, const float* __restrict__ Sw,
                   float* __restrict__ out, int D, int N) {
    __shared__ __align__(16) float xbuf[2][TT][LSTR];
    __shared__ __align__(16) float wbuf[2][TJ][LSTR];

    const int tid  = threadIdx.x;
    const int tr0  = blockIdx.x * TT;
    const int jc0  = blockIdx.y * TJ;
    const int tloc = (tid >> 4) * 4;
    const int jloc = (tid & 15) * 4;

    float acc[4][4] = {};

    const int nkb = D / KC;

    // Prologue: DMA first tiles into buffer 0 (wave 0 only; EXEC ignored by TDM)
    if (tid < 32) {
        tdm_load_tile(lds_off_of(&xbuf[0][0][0]), X + (size_t)tr0 * D, (unsigned)D);
        tdm_load_tile(lds_off_of(&wbuf[0][0][0]), W + (size_t)jc0 * D, (unsigned)D);
    }

    for (int kb = 0; kb < nkb; ++kb) {
        const int cur = kb & 1;
        const int nxt = cur ^ 1;

        if (tid < 32) {
            __builtin_amdgcn_s_wait_tensorcnt(0);   // current buffer landed
        }
        __syncthreads();   // release buffer to all waves; also proves everyone
                           // finished reading the buffer we are about to refill

        if (tid < 32 && (kb + 1) < nkb) {
            tdm_load_tile(lds_off_of(&xbuf[nxt][0][0]),
                          X + (size_t)tr0 * D + (size_t)(kb + 1) * KC, (unsigned)D);
            tdm_load_tile(lds_off_of(&wbuf[nxt][0][0]),
                          W + (size_t)jc0 * D + (size_t)(kb + 1) * KC, (unsigned)D);
        }

        // ---- min-sum micro-kernel: 8x ds_load_b128 per 128 VALU per k-quad ----
#pragma unroll
        for (int kq = 0; kq < KC / 4; ++kq) {
            float4 xq[4], wq[4];
#pragma unroll
            for (int a = 0; a < 4; ++a)
                xq[a] = *(const float4*)&xbuf[cur][tloc + a][4 * kq];
#pragma unroll
            for (int b = 0; b < 4; ++b)
                wq[b] = *(const float4*)&wbuf[cur][jloc + b][4 * kq];
#pragma unroll
            for (int a = 0; a < 4; ++a)
#pragma unroll
                for (int b = 0; b < 4; ++b) {
                    acc[a][b] += fminf(xq[a].x, wq[b].x);
                    acc[a][b] += fminf(xq[a].y, wq[b].y);
                    acc[a][b] += fminf(xq[a].z, wq[b].z);
                    acc[a][b] += fminf(xq[a].w, wq[b].w);
                }
        }
        __syncthreads();
    }

    // ---- epilogue ----
    const float invD = 1.0f / (float)D;
#pragma unroll
    for (int a = 0; a < 4; ++a) {
        const int t = tr0 + tloc + a;
        const float sx = Sx[t];
        float res[4];
#pragma unroll
        for (int b = 0; b < 4; ++b) {
            const int j = jc0 + jloc + b;
            const float s1 = sx + Sw[j] - 2.0f * acc[a][b];
            float v = bias[j] - s1 * invD;
            if (SPIKE)
                v = 1.0f / (1.0f + __expf(-4.0f * (v - vth[j])));
            res[b] = v;
        }
        float4 o = {res[0], res[1], res[2], res[3]};
        *(float4*)(out + (size_t)t * N + jc0 + jloc) = o;
    }
}

// ---------------------------------------------------------------------------
// B=2, S=512 -> T=1024; D_MODEL=512; MAX_NEURONS=2048.
// d_in order: x, W_in, b_in, W_out, b_out, v_th (all f32)
// ---------------------------------------------------------------------------
extern "C" void kernel_launch(void* const* d_in, const int* in_sizes, int n_in,
                              void* d_out, int out_size, void* d_ws, size_t ws_size,
                              hipStream_t stream) {
    (void)in_sizes; (void)n_in; (void)out_size; (void)ws_size;

    const float* x     = (const float*)d_in[0];
    const float* W_in  = (const float*)d_in[1];
    const float* b_in  = (const float*)d_in[2];
    const float* W_out = (const float*)d_in[3];
    const float* b_out = (const float*)d_in[4];
    const float* v_th  = (const float*)d_in[5];
    float* out = (float*)d_out;

    const int T = 1024, D = 512, NN = 2048;

    float* ws     = (float*)d_ws;
    float* spikes = ws;                       // [1024 x 2048]
    float* Sx1    = spikes + (size_t)T * NN;  // [1024]
    float* Sw1    = Sx1 + T;                  // [2048]
    float* Ssp    = Sw1 + NN;                 // [1024]
    float* Sw2    = Ssp + T;                  // [512]

    rowsum_wmma_kernel<<<T  / 128, 256, 0, stream>>>(x,     D,  Sx1);
    rowsum_wmma_kernel<<<NN / 128, 256, 0, stream>>>(W_in,  D,  Sw1);

    addmin_kernel<true><<<dim3(T / TT, NN / TJ), 256, 0, stream>>>(
        x, W_in, b_in, v_th, Sx1, Sw1, spikes, D, NN);

    rowsum_wmma_kernel<<<T / 128, 256, 0, stream>>>(spikes, NN, Ssp);
    rowsum_wmma_kernel<<<D / 128, 256, 0, stream>>>(W_out,  NN, Sw2);

    addmin_kernel<false><<<dim3(T / TT, D / TJ), 256, 0, stream>>>(
        spikes, W_out, b_out, nullptr, Ssp, Sw2, out, NN, D);
}
